// LeftKanExtension_35820027249104
// MI455X (gfx1250) — matrix operand order repaired and checked by reading, not compiled, not run on hardware
//
#include <hip/hip_runtime.h>
#include <stdint.h>

#define FDIM   128
#define NQ     4096
#define NK     4096
#define BATCH  4
#define MTILE  128    // query rows per block (8 waves x 16)
#define KTILE  64     // keys per main-loop iteration
#define NWAVES 8
#define NITER  (NK / KTILE)

typedef __bf16 bf16;
typedef __attribute__((ext_vector_type(16))) __bf16 bf16x16;
typedef __attribute__((ext_vector_type(8)))  __bf16 bf16x8;
typedef __attribute__((ext_vector_type(8)))  float  f32x8;
typedef __attribute__((ext_vector_type(4)))  float  f32x4;

// padded LDS pitches (elements) to spread banks on row-strided b128 reads
#define LK_PITCH  136   // K tile rows: 128 features + 8 pad  (68 dword banks/row)
#define LVT_PITCH 72    // V^T rows:    64 keys + 8 pad       (36 dword banks/row)
#define LP_PITCH  72    // P rows:      64 keys + 8 pad

__global__ __launch_bounds__(256)
void flash_attn_f32_bf16wmma(const float* __restrict__ Q,
                             const float* __restrict__ K,
                             const float* __restrict__ V,
                             float* __restrict__ Out)
{
    __shared__ float rawK[KTILE * FDIM];            // async-staged fp32 K tile (32 KB)
    __shared__ float rawV[KTILE * FDIM];            // async-staged fp32 V tile (32 KB)
    __shared__ bf16  lK [KTILE * LK_PITCH];         // [key][feature]  bf16
    __shared__ bf16  lVt[FDIM  * LVT_PITCH];        // [feature][key]  bf16 (transposed V)
    __shared__ bf16  lP [NWAVES * 16 * LP_PITCH];   // per-wave P staging [row][key]

    const int tid  = threadIdx.x;
    const int wave = tid >> 5;
    const int lane = tid & 31;
    const int h    = lane >> 4;   // half-wave select
    const int n    = lane & 15;   // column / row-in-frag index

    const int blk   = blockIdx.x;
    const int batch = blk / (NQ / MTILE);
    const int qtile = blk % (NQ / MTILE);
    const int qrow0 = qtile * MTILE + wave * 16;     // this wave's first query row

    // 1/sqrt(128) * log2(e): scores land directly in base-2 softmax domain
    const float scale2 = 0.08838834764831845f * 1.4426950408889634f;

    // ---- preload Q as 4 pre-scaled bf16 A-fragments (16 x 32 each) ----
    bf16x16 aq[4];
    {
        const float* qp = Q + ((size_t)batch * NQ + qrow0 + n) * FDIM;
#pragma unroll
        for (int kk = 0; kk < 4; ++kk) {
            const int c0 = kk * 32 + h * 8;
            f32x4 f0 = *(const f32x4*)(qp + c0);
            f32x4 f1 = *(const f32x4*)(qp + c0 + 4);
            f32x4 f2 = *(const f32x4*)(qp + c0 + 16);
            f32x4 f3 = *(const f32x4*)(qp + c0 + 20);
            bf16x16 a;
#pragma unroll
            for (int e = 0; e < 4; ++e) {
                a[e]      = (bf16)(f0[e] * scale2);
                a[4 + e]  = (bf16)(f1[e] * scale2);
                a[8 + e]  = (bf16)(f2[e] * scale2);
                a[12 + e] = (bf16)(f3[e] * scale2);
            }
            aq[kk] = a;
        }
    }

    // ---- accumulators + online-softmax state (rows M = 8*h + i), base-2 domain ----
    f32x8 acc[8];
#pragma unroll
    for (int t = 0; t < 8; ++t) acc[t] = (f32x8){0.f,0.f,0.f,0.f,0.f,0.f,0.f,0.f};
    float mrow[8], lrow[8];
#pragma unroll
    for (int i = 0; i < 8; ++i) { mrow[i] = -3.0e38f; lrow[i] = 0.f; }

    // ---- async staging mapping: thread -> (key row, 32-float column chunk) ----
    const int lrowi = tid >> 2;           // 0..63
    const int lc0   = (tid & 3) * 32;     // 0..96
    const float* kg0 = K + ((size_t)batch * NK + lrowi) * FDIM + lc0;
    const float* vg0 = V + ((size_t)batch * NK + lrowi) * FDIM + lc0;
    const unsigned rawKoff = (unsigned)(uintptr_t)&rawK[lrowi * FDIM + lc0];
    const unsigned rawVoff = (unsigned)(uintptr_t)&rawV[lrowi * FDIM + lc0];

    // fire one tile's K/V into LDS via the async DMA path (ASYNCcnt-tracked)
    auto issue_tile = [&](int kt) {
        const float* kg = kg0 + (size_t)kt * KTILE * FDIM;
        const float* vg = vg0 + (size_t)kt * KTILE * FDIM;
#pragma unroll
        for (int m = 0; m < 8; ++m) {
            asm volatile("global_load_async_to_lds_b128 %0, %1, off"
                         :: "v"(rawKoff + 16u * m), "v"(kg + 4 * m) : "memory");
            asm volatile("global_load_async_to_lds_b128 %0, %1, off"
                         :: "v"(rawVoff + 16u * m), "v"(vg + 4 * m) : "memory");
        }
    };

    issue_tile(0);

    for (int kt = 0; kt < NITER; ++kt) {
        // wait for this wave's async writes, then publish to all waves
        asm volatile("s_wait_asynccnt 0" ::: "memory");
        __syncthreads();

        // ---- convert raw fp32 tile -> bf16 lK (row-major) and lVt (transposed) ----
#pragma unroll
        for (int m = 0; m < 8; ++m) {
            f32x4 kf = *(const f32x4*)&rawK[lrowi * FDIM + lc0 + 4 * m];
            union { bf16 b[4]; unsigned long long u; } pk;
            pk.b[0] = (bf16)kf[0]; pk.b[1] = (bf16)kf[1];
            pk.b[2] = (bf16)kf[2]; pk.b[3] = (bf16)kf[3];
            *(unsigned long long*)&lK[lrowi * LK_PITCH + lc0 + 4 * m] = pk.u;

            f32x4 vf = *(const f32x4*)&rawV[lrowi * FDIM + lc0 + 4 * m];
            lVt[(lc0 + 4 * m + 0) * LVT_PITCH + lrowi] = (bf16)vf[0];
            lVt[(lc0 + 4 * m + 1) * LVT_PITCH + lrowi] = (bf16)vf[1];
            lVt[(lc0 + 4 * m + 2) * LVT_PITCH + lrowi] = (bf16)vf[2];
            lVt[(lc0 + 4 * m + 3) * LVT_PITCH + lrowi] = (bf16)vf[3];
        }
        __syncthreads();   // raw buffers drained; bf16 tiles visible

        // overlap: refill raw buffers for the next tile while we compute this one
        if (kt + 1 < NITER) issue_tile(kt + 1);

        // ---- S = (Q*scale2) * K^T : this wave's 16 rows vs 64 keys ----
        float s[4][8];
#pragma unroll
        for (int jn = 0; jn < 4; ++jn) {
            f32x8 c = (f32x8){0.f,0.f,0.f,0.f,0.f,0.f,0.f,0.f};
            const bf16* kr = &lK[(jn * 16 + n) * LK_PITCH];
#pragma unroll
            for (int kk = 0; kk < 4; ++kk) {
                union { bf16x16 v; bf16x8 half[2]; } bf;
                bf.half[0] = *(const bf16x8*)(kr + kk * 32 + h * 8);
                bf.half[1] = *(const bf16x8*)(kr + kk * 32 + h * 8 + 16);
                c = __builtin_amdgcn_wmma_f32_16x16x32_bf16(
                        false, aq[kk], false, bf.v, (short)0, c, false, false);
            }
#pragma unroll
            for (int i = 0; i < 8; ++i) s[jn][i] = c[i];
        }

        // ---- online softmax update (base-2; reductions across 16-lane halves) ----
        float mnew[8], alpha[8], rsum[8];
#pragma unroll
        for (int i = 0; i < 8; ++i) {
            float rm = fmaxf(fmaxf(s[0][i], s[1][i]), fmaxf(s[2][i], s[3][i]));
            rm = fmaxf(rm, __shfl_xor(rm, 1, 32));
            rm = fmaxf(rm, __shfl_xor(rm, 2, 32));
            rm = fmaxf(rm, __shfl_xor(rm, 4, 32));
            rm = fmaxf(rm, __shfl_xor(rm, 8, 32));
            mnew[i]  = fmaxf(mrow[i], rm);
            alpha[i] = exp2f(mrow[i] - mnew[i]);
            rsum[i]  = 0.f;
        }

        bf16* pst = &lP[(wave * 16) * LP_PITCH];
#pragma unroll
        for (int jn = 0; jn < 4; ++jn) {
#pragma unroll
            for (int i = 0; i < 8; ++i) {
                float p = exp2f(s[jn][i] - mnew[i]);
                rsum[i] += p;
                pst[(8 * h + i) * LP_PITCH + jn * 16 + n] = (bf16)p;
            }
        }
#pragma unroll
        for (int i = 0; i < 8; ++i) {
            float r = rsum[i];
            r += __shfl_xor(r, 1, 32);
            r += __shfl_xor(r, 2, 32);
            r += __shfl_xor(r, 4, 32);
            r += __shfl_xor(r, 8, 32);
            lrow[i] = lrow[i] * alpha[i] + r;
            mrow[i] = mnew[i];
        }
        // unconditional rescale: branch-free, dual-issue muls co-execute with XDL
#pragma unroll
        for (int t = 0; t < 8; ++t)
#pragma unroll
            for (int i = 0; i < 8; ++i) acc[t][i] *= alpha[i];

        // same-wave LDS store -> load fence for the P staging buffer
        asm volatile("s_wait_dscnt 0" ::: "memory");

        // ---- O += P * V  (K = 64 keys as two k=32 passes, 8 d-tiles) ----
#pragma unroll
        for (int kq = 0; kq < 2; ++kq) {
            union { bf16x16 v; bf16x8 half[2]; } pf;
            const bf16* pr = &lP[(wave * 16 + n) * LP_PITCH + kq * 32 + 8 * h];
            pf.half[0] = *(const bf16x8*)(pr);
            pf.half[1] = *(const bf16x8*)(pr + 16);
#pragma unroll
            for (int dt = 0; dt < 8; ++dt) {
                union { bf16x16 v; bf16x8 half[2]; } vf;
                const bf16* vr = &lVt[(dt * 16 + n) * LVT_PITCH + kq * 32 + 8 * h];
                vf.half[0] = *(const bf16x8*)(vr);
                vf.half[1] = *(const bf16x8*)(vr + 16);
                acc[dt] = __builtin_amdgcn_wmma_f32_16x16x32_bf16(
                              false, pf.v, false, vf.v, (short)0, acc[dt], false, false);
            }
        }
    }

    // ---- normalize and store (C layout: VGPR i -> row 8h+i, col = dt*16+n) ----
    float inv[8];
#pragma unroll
    for (int i = 0; i < 8; ++i) inv[i] = 1.0f / lrow[i];
    float* op = Out + ((size_t)batch * NQ + qrow0) * FDIM;
#pragma unroll
    for (int dt = 0; dt < 8; ++dt)
#pragma unroll
        for (int i = 0; i < 8; ++i)
            op[(size_t)(8 * h + i) * FDIM + dt * 16 + n] = acc[dt][i] * inv[i];
}

extern "C" void kernel_launch(void* const* d_in, const int* in_sizes, int n_in,
                              void* d_out, int out_size, void* d_ws, size_t ws_size,
                              hipStream_t stream) {
    (void)in_sizes; (void)n_in; (void)out_size; (void)d_ws; (void)ws_size;
    const float* Q = (const float*)d_in[0];
    const float* K = (const float*)d_in[1];
    const float* V = (const float*)d_in[2];
    float* O = (float*)d_out;
    dim3 grid(BATCH * (NQ / MTILE));
    dim3 block(256);
    hipLaunchKernelGGL(flash_attn_f32_bf16wmma, grid, block, 0, stream, Q, K, V, O);
}
